// MultiHeadAttention_44976897524050
// MI455X (gfx1250) — matrix operand order repaired and checked by reading, not compiled, not run on hardware
//
#include <hip/hip_runtime.h>

// ---------------------------------------------------------------------------
// MultiHeadAttention forward for MI455X (gfx1250, wave32, WMMA bf16)
//   out  = (ctx @ Wo^T + bo)                 [B,S,1024] fp32
//   attn = softmax(mask(scale * Q K^T))      [B,H,S,S]  fp32
// Pipeline: fp32->bf16 convert -> 3 WMMA proj GEMMs -> fused
// scores/softmax/PV kernel (128KB LDS row-block) -> WMMA output GEMM.
// Round-3: round-1 loop structure (compiler-scheduled, fully unrolled WMMA
// clauses; latency hidden by multi-wave occupancy) + non-temporal stores for
// the write-once attn/out streams + float4 softmax path + NT convert loads.
// ---------------------------------------------------------------------------

typedef __attribute__((ext_vector_type(16))) __bf16 v16bf;
typedef __attribute__((ext_vector_type(8)))  __bf16 v8bf;
typedef __attribute__((ext_vector_type(4)))  __bf16 v4bf;
typedef __attribute__((ext_vector_type(8)))  float  v8f;
typedef __attribute__((ext_vector_type(4)))  float  v4f;

static constexpr int SQ = 2048;   // sequence length
static constexpr int DM = 1024;   // d_model
static constexpr int NH = 16;     // heads
static constexpr int DK = 64;     // head dim
static constexpr int BB = 2;      // batch
static constexpr int MROWS = BB * SQ;      // 4096
static constexpr float SCALE = 0.125f;     // 1/sqrt(64)
static constexpr float NEGINF = -1e10f;

__device__ inline v8f vzero8() {
  v8f z = {0.f,0.f,0.f,0.f,0.f,0.f,0.f,0.f};
  return z;
}

// A fragment (16x32 bf16, M x K) from row-major memory, K contiguous, stride ld.
// ISA layout: lane m (0..15): elems 0..7 = A[m][hh*8 .. +7], elems 8..15 =
// A[m][16+hh*8 .. +7], where hh = lane>>4.
__device__ inline v16bf load_a_bf16(const __bf16* __restrict__ A, int ld) {
  const int l  = threadIdx.x & 31;
  const int m  = l & 15;
  const int hh = l >> 4;
  union { v16bf v; v8bf h[2]; } u;
  const __bf16* p = A + (size_t)m * ld + hh * 8;
  u.h[0] = *(const v8bf*)(p);
  u.h[1] = *(const v8bf*)(p + 16);
  return u.v;
}

// B fragment (32x16 bf16, K x N) where the source stores B^T row-major
// (rows = N, K contiguous, stride ld). ISA layout: lane n (0..15): elems 0..7 =
// K=hh*8..+7, elems 8..15 = K=16+hh*8..+7 of column n.
__device__ inline v16bf load_b_bf16(const __bf16* __restrict__ Bt, int ld) {
  const int l  = threadIdx.x & 31;
  const int n  = l & 15;
  const int hh = l >> 4;
  union { v16bf v; v8bf h[2]; } u;
  const __bf16* p = Bt + (size_t)n * ld + hh * 8;
  u.h[0] = *(const v8bf*)(p);
  u.h[1] = *(const v8bf*)(p + 16);
  return u.v;
}

// A fragment assembled from fp32 LDS data (softmax probabilities).
__device__ inline v16bf load_a_f32(const float* __restrict__ A, int ld) {
  const int l  = threadIdx.x & 31;
  const int m  = l & 15;
  const int hh = l >> 4;
  const float* p0 = A + (size_t)m * ld + hh * 8;
  const float* p1 = p0 + 16;
  v4f f0 = *(const v4f*)(p0);
  v4f f1 = *(const v4f*)(p0 + 4);
  v4f f2 = *(const v4f*)(p1);
  v4f f3 = *(const v4f*)(p1 + 4);
  union { v16bf v; __bf16 e[16]; } u;
#pragma unroll
  for (int j = 0; j < 4; ++j) {
    u.e[j]      = (__bf16)f0[j];
    u.e[4 + j]  = (__bf16)f1[j];
    u.e[8 + j]  = (__bf16)f2[j];
    u.e[12 + j] = (__bf16)f3[j];
  }
  return u.v;
}

// ---------------------------------------------------------------------------
// fp32 -> bf16 convert (vectorized x4). Inputs are read exactly once ->
// non-temporal loads keep them out of L2; destinations are reused -> RT store.
// ---------------------------------------------------------------------------
__global__ void cvt_f32_bf16(const v4f* __restrict__ src,
                             v4bf* __restrict__ dst, int n4) {
  int i = blockIdx.x * blockDim.x + threadIdx.x;
  int stride = gridDim.x * blockDim.x;
  for (; i < n4; i += stride) {
    v4f f = __builtin_nontemporal_load(src + i);
    v4bf o;
    o[0] = (__bf16)f[0]; o[1] = (__bf16)f[1]; o[2] = (__bf16)f[2]; o[3] = (__bf16)f[3];
    dst[i] = o;
  }
}

// ---------------------------------------------------------------------------
// C = X @ W^T + bias.  X: (M x Kd) bf16 row-major, W: (N x Kd) bf16 row-major.
// MODE 0: bf16 out row-major (M x N)
// MODE 1: bf16 out as V-transposed (B,H,DK,S): vt[b][h][c][s] = C[b*S+s][h*64+c]
// MODE 2: fp32 out row-major (M x N), non-temporal (final result, never re-read)
// One wave computes a 16x64 output tile (A-frag reuse over 4 N-tiles).
// ---------------------------------------------------------------------------
template <int MODE>
__global__ __launch_bounds__(128) void gemm_xwT(
    const __bf16* __restrict__ X, const __bf16* __restrict__ W,
    const float* __restrict__ bias, float* __restrict__ outF,
    __bf16* __restrict__ outB, int M, int N, int Kd) {
  const int tilesN = N / 64;
  const int wave = blockIdx.x * (blockDim.x >> 5) + (threadIdx.x >> 5);
  const int mt = wave / tilesN;
  const int nt = wave % tilesN;
  if (mt * 16 >= M) return;

  const int l   = threadIdx.x & 31;
  const int hh  = l >> 4;
  const int n16 = l & 15;

  v8f acc[4] = {vzero8(), vzero8(), vzero8(), vzero8()};

  const __bf16* Xt = X + (size_t)mt * 16 * Kd;
  const __bf16* Wt = W + (size_t)nt * 64 * Kd;

  for (int k0 = 0; k0 < Kd; k0 += 32) {
    v16bf a = load_a_bf16(Xt + k0, Kd);
#pragma unroll
    for (int j = 0; j < 4; ++j) {
      v16bf b = load_b_bf16(Wt + (size_t)j * 16 * Kd + k0, Kd);
      acc[j] = __builtin_amdgcn_wmma_f32_16x16x32_bf16(
          false, a, false, b, (short)0, acc[j], false, false);
    }
  }

#pragma unroll
  for (int j = 0; j < 4; ++j) {
#pragma unroll
    for (int e = 0; e < 8; ++e) {
      const int row = mt * 16 + e + 8 * hh;
      const int col = nt * 64 + j * 16 + n16;
      const float v = acc[j][e] + bias[col];
      if (MODE == 2) {
        __builtin_nontemporal_store(v, &outF[(size_t)row * N + col]);
      } else if (MODE == 0) {
        outB[(size_t)row * N + col] = (__bf16)v;
      } else {  // MODE 1: V transposed per head
        const int b = row >> 11;      // row / S
        const int s = row & (SQ - 1);
        const int h = col >> 6;       // col / DK
        const int c = col & (DK - 1);
        outB[(((size_t)b * NH + h) * DK + c) * SQ + s] = (__bf16)v;
      }
    }
  }
}

// ---------------------------------------------------------------------------
// Fused attention per (b, h, 16-query-row tile):
//   scores (WMMA) -> mask -> softmax (LDS) -> write attn (NT) -> P@V (WMMA)
// 256 threads (8 waves). LDS: probs 16x2048 f32 (128KB) + partials 8x16x64 f32.
// 160KB/block -> 2 blocks (16 waves) per WGP; 4 waves/SIMD hide load latency.
// ---------------------------------------------------------------------------
__global__ __launch_bounds__(256) void attn_fused(
    const __bf16* __restrict__ Qp,   // (B,S,DM) bf16
    const __bf16* __restrict__ Kp,   // (B,S,DM) bf16
    const __bf16* __restrict__ Vt,   // (B,H,DK,S) bf16
    const int* __restrict__ mask,    // (B,S) key_padding_mask (nonzero = keep)
    float* __restrict__ attn,        // (B,H,S,S) fp32
    __bf16* __restrict__ ctx) {      // (B,S,DM) bf16
  extern __shared__ float smem[];
  float* probs = smem;                  // 16 * SQ
  float* part  = smem + 16 * SQ;        // 8 * 16 * 64

  const int blk = blockIdx.x;
  const int qt  = blk & 127;
  const int h   = (blk >> 7) & 15;
  const int b   = blk >> 11;
  const int w   = threadIdx.x >> 5;
  const int l   = threadIdx.x & 31;
  const int hh  = l >> 4;
  const int n16 = l & 15;
  const int q0  = qt * 16;

  // ---- phase 1: masked scaled logits into LDS ----
  const __bf16* Qbase = Qp + ((size_t)(b * SQ + q0)) * DM + h * DK;
  v16bf a0 = load_a_bf16(Qbase + 0, DM);
  v16bf a1 = load_a_bf16(Qbase + 32, DM);

#pragma unroll
  for (int t = 0; t < 16; ++t) {
    const int c0 = w * 256 + t * 16;
    const __bf16* Kbase = Kp + ((size_t)(b * SQ + c0)) * DM + h * DK;
    v16bf b0 = load_b_bf16(Kbase + 0, DM);
    v16bf b1 = load_b_bf16(Kbase + 32, DM);
    v8f s = vzero8();
    s = __builtin_amdgcn_wmma_f32_16x16x32_bf16(false, a0, false, b0, (short)0, s, false, false);
    s = __builtin_amdgcn_wmma_f32_16x16x32_bf16(false, a1, false, b1, (short)0, s, false, false);
    const int c = c0 + n16;
    const bool keyok = (mask[b * SQ + c] != 0);
#pragma unroll
    for (int e = 0; e < 8; ++e) {
      const int lr = e + 8 * hh;
      const int qg = q0 + lr;
      float v = s[e] * SCALE;
      if (!keyok || c > qg) v = NEGINF;
      probs[lr * SQ + c] = v;
    }
  }
  __syncthreads();

  // ---- phase 2: softmax over each row (2 rows per wave, 16 lanes per row,
  //      float4 strides; attn written once, non-temporal b128 stores) ----
  {
    const int lr = 2 * w + hh;        // 0..15
    float* prow = probs + lr * SQ;
    float m = -3.4e38f;
    for (int i = n16; i < SQ / 4; i += 16) {
      v4f f = *(const v4f*)(prow + i * 4);
      m = fmaxf(m, fmaxf(fmaxf(f[0], f[1]), fmaxf(f[2], f[3])));
    }
#pragma unroll
    for (int off = 1; off < 16; off <<= 1) m = fmaxf(m, __shfl_xor(m, off, 32));
    float sum = 0.f;
    for (int i = n16; i < SQ / 4; i += 16) {
      v4f f = *(const v4f*)(prow + i * 4);
      v4f e;
      e[0] = __expf(f[0] - m); e[1] = __expf(f[1] - m);
      e[2] = __expf(f[2] - m); e[3] = __expf(f[3] - m);
      *(v4f*)(prow + i * 4) = e;
      sum += e[0] + e[1] + e[2] + e[3];
    }
#pragma unroll
    for (int off = 1; off < 16; off <<= 1) sum += __shfl_xor(sum, off, 32);
    const float inv = 1.f / sum;
    float* arow = attn + (((size_t)(b * NH + h)) * SQ + (q0 + lr)) * SQ;
    for (int i = n16; i < SQ / 4; i += 16) {
      v4f f = *(const v4f*)(prow + i * 4);
      f[0] *= inv; f[1] *= inv; f[2] *= inv; f[3] *= inv;
      *(v4f*)(prow + i * 4) = f;
      __builtin_nontemporal_store(f, (v4f*)(arow + i * 4));  // one-shot HBM write
    }
  }
  __syncthreads();

  // ---- phase 3: ctx_tile(16x64) = P(16x2048) @ V_h(2048x64), split-K over waves
  v8f acc[4] = {vzero8(), vzero8(), vzero8(), vzero8()};
  const __bf16* Vbase = Vt + ((size_t)(b * NH + h)) * DK * SQ;  // rows=c, ld=SQ
#pragma unroll
  for (int tk = 0; tk < 8; ++tk) {
    const int kk = w * 256 + tk * 32;
    v16bf a = load_a_f32(probs + kk, SQ);
#pragma unroll
    for (int j = 0; j < 4; ++j) {
      v16bf bb = load_b_bf16(Vbase + (size_t)(j * 16) * SQ + kk, SQ);
      acc[j] = __builtin_amdgcn_wmma_f32_16x16x32_bf16(
          false, a, false, bb, (short)0, acc[j], false, false);
    }
  }

#pragma unroll
  for (int j = 0; j < 4; ++j) {
#pragma unroll
    for (int e = 0; e < 8; ++e) {
      const int lr = e + 8 * hh;
      part[(w * 16 + lr) * 64 + j * 16 + n16] = acc[j][e];
    }
  }
  __syncthreads();

  for (int o = threadIdx.x; o < 16 * 64; o += 256) {
    const int lr = o >> 6;
    const int c  = o & 63;
    float s = 0.f;
#pragma unroll
    for (int p = 0; p < 8; ++p) s += part[(p * 16 + lr) * 64 + c];
    ctx[((size_t)(b * SQ + q0 + lr)) * DM + h * DK + c] = (__bf16)s;
  }
}

// ---------------------------------------------------------------------------
// Host-side launcher
// ---------------------------------------------------------------------------
extern "C" void kernel_launch(void* const* d_in, const int* in_sizes, int n_in,
                              void* d_out, int out_size, void* d_ws, size_t ws_size,
                              hipStream_t stream) {
  const float* q_in = (const float*)d_in[0];
  const float* k_in = (const float*)d_in[1];
  const float* v_in = (const float*)d_in[2];
  const int*   mask = (const int*)d_in[3];
  const float* Wq = (const float*)d_in[4];  const float* bq = (const float*)d_in[5];
  const float* Wk = (const float*)d_in[6];  const float* bk = (const float*)d_in[7];
  const float* Wv = (const float*)d_in[8];  const float* bv = (const float*)d_in[9];
  const float* Wo = (const float*)d_in[10]; const float* bo = (const float*)d_in[11];

  float* out  = (float*)d_out;
  float* attn = out + (size_t)BB * SQ * DM;   // out: 4.19M floats, then attn

  const size_t nAct = (size_t)MROWS * DM;     // 4096*1024
  const size_t nW   = (size_t)DM * DM;        // 1024*1024

  char* ws = (char*)d_ws;
  __bf16* xq  = (__bf16*)ws; ws += nAct * 2;
  __bf16* xk  = (__bf16*)ws; ws += nAct * 2;
  __bf16* xv  = (__bf16*)ws; ws += nAct * 2;
  __bf16* wqb = (__bf16*)ws; ws += nW * 2;
  __bf16* wkb = (__bf16*)ws; ws += nW * 2;
  __bf16* wvb = (__bf16*)ws; ws += nW * 2;
  __bf16* wob = (__bf16*)ws; ws += nW * 2;
  __bf16* qp  = (__bf16*)ws; ws += nAct * 2;
  __bf16* kp  = (__bf16*)ws; ws += nAct * 2;
  __bf16* vt  = (__bf16*)ws; ws += nAct * 2;   // (B,H,DK,S)
  __bf16* ctx = (__bf16*)ws; ws += nAct * 2;

  // 1) fp32 -> bf16 conversions
  {
    const int nAct4 = (int)(nAct / 4), nW4 = (int)(nW / 4);
    dim3 blk(256);
    cvt_f32_bf16<<<(nAct4 + 255) / 256, blk, 0, stream>>>((const v4f*)q_in, (v4bf*)xq,  nAct4);
    cvt_f32_bf16<<<(nAct4 + 255) / 256, blk, 0, stream>>>((const v4f*)k_in, (v4bf*)xk,  nAct4);
    cvt_f32_bf16<<<(nAct4 + 255) / 256, blk, 0, stream>>>((const v4f*)v_in, (v4bf*)xv,  nAct4);
    cvt_f32_bf16<<<(nW4   + 255) / 256, blk, 0, stream>>>((const v4f*)Wq,   (v4bf*)wqb, nW4);
    cvt_f32_bf16<<<(nW4   + 255) / 256, blk, 0, stream>>>((const v4f*)Wk,   (v4bf*)wkb, nW4);
    cvt_f32_bf16<<<(nW4   + 255) / 256, blk, 0, stream>>>((const v4f*)Wv,   (v4bf*)wvb, nW4);
    cvt_f32_bf16<<<(nW4   + 255) / 256, blk, 0, stream>>>((const v4f*)Wo,   (v4bf*)wob, nW4);
  }

  // 2) projections: (4096/16)*(1024/64) = 4096 wave-tiles / 4 waves per block
  const int gemmBlocks = (MROWS / 16) * (DM / 64) / 4;  // 1024
  gemm_xwT<0><<<gemmBlocks, 128, 0, stream>>>(xq, wqb, bq, nullptr, qp, MROWS, DM, DM);
  gemm_xwT<0><<<gemmBlocks, 128, 0, stream>>>(xk, wkb, bk, nullptr, kp, MROWS, DM, DM);
  gemm_xwT<1><<<gemmBlocks, 128, 0, stream>>>(xv, wvb, bv, nullptr, vt, MROWS, DM, DM);

  // 3) fused scores/softmax/PV: B*H*(S/16) = 4096 blocks, 160KB dynamic LDS
  {
    const size_t shmem = (size_t)(16 * SQ + 8 * 16 * 64) * sizeof(float);  // 163840
    attn_fused<<<BB * NH * (SQ / 16), 256, shmem, stream>>>(qp, kp, vt, mask, attn, ctx);
  }

  // 4) output projection (fp32 result, non-temporal stores)
  gemm_xwT<2><<<gemmBlocks, 128, 0, stream>>>(ctx, wob, bo, out, nullptr, MROWS, DM, DM);
}
